// DirectionalStockGNN_62672162783774
// MI455X (gfx1250) — matrix operand order repaired and checked by reading, not compiled, not run on hardware
//
#include <hip/hip_runtime.h>
#include <hip/hip_bf16.h>
#include <math.h>

#define DH 128
#define NEG_SLOPE 0.2f

typedef __attribute__((ext_vector_type(2))) float v2f;
typedef __attribute__((ext_vector_type(8))) float v8f;
#if !__has_builtin(__builtin_amdgcn_wmma_f32_16x16x4_f32)
typedef __attribute__((ext_vector_type(16))) __bf16 v16bf;
#endif

// ---- order-preserving float<->uint encoding for atomic segment-max ----
__device__ __forceinline__ unsigned enc_f32(float f) {
  unsigned u = __float_as_uint(f);
  return (u & 0x80000000u) ? ~u : (u | 0x80000000u);
}
__device__ __forceinline__ float dec_f32(unsigned u) {
  unsigned b = (u & 0x80000000u) ? (u ^ 0x80000000u) : ~u;
  return __uint_as_float(b);
}

__device__ __forceinline__ float wave_sum32(float v) {
  #pragma unroll
  for (int off = 16; off > 0; off >>= 1) v += __shfl_xor(v, off, 32);
  return v;
}

// =====================================================================
// WMMA GEMM: OUT[nTiles*16, 128] = X[nTiles*16, 128] @ W[128, 128]
// One wave computes a 16-row x 128-col strip. fp32-exact via 16x16x4_f32
// (bf16 fallback if the f32 builtin is absent on this toolchain).
// =====================================================================
__global__ __launch_bounds__(256) void gemm_x128_wmma(
    const float* __restrict__ X, const float* __restrict__ W,
    float* __restrict__ OUT, int nTiles) {
  const int lane = threadIdx.x & 31;
  const int tile = (int)((blockIdx.x * blockDim.x + threadIdx.x) >> 5);
  if (tile >= nTiles) return;               // wave-uniform
  const int half = lane >> 4;
  const int l16  = lane & 15;
  const long row = (long)tile * 16 + l16;
  const float* __restrict__ xrow = X + row * DH;

  v8f c[8];
  #pragma unroll
  for (int j = 0; j < 8; ++j)
    #pragma unroll
    for (int i = 0; i < 8; ++i) c[j][i] = 0.0f;

#if __has_builtin(__builtin_amdgcn_wmma_f32_16x16x4_f32)
  for (int k0 = 0; k0 < DH; k0 += 4) {
    const int k = k0 + 2 * half;            // A: VGPR j holds K = 2*half + j
    v2f a;
    a.x = xrow[k];
    a.y = xrow[k + 1];
    #pragma unroll
    for (int j = 0; j < 8; ++j) {
      v2f b;                                 // B mirrors A's K split
      b.x = W[(size_t)k * DH + j * 16 + l16];
      b.y = W[(size_t)(k + 1) * DH + j * 16 + l16];
      c[j] = __builtin_amdgcn_wmma_f32_16x16x4_f32(
          false, a, false, b, (short)0, c[j], false, false);
    }
  }
#else
  for (int k0 = 0; k0 < DH; k0 += 32) {
    v16bf a;
    #pragma unroll
    for (int t = 0; t < 16; ++t) {
      int k = k0 + 16 * half + t;
      a[t] = (__bf16)xrow[k];
    }
    #pragma unroll
    for (int j = 0; j < 8; ++j) {
      v16bf b;
      #pragma unroll
      for (int t = 0; t < 16; ++t)
        b[t] = (__bf16)W[(size_t)(k0 + 16 * half + t) * DH + j * 16 + l16];
      c[j] = __builtin_amdgcn_wmma_f32_16x16x32_bf16(
          false, a, false, b, (short)0, c[j], false, false);
    }
  }
#endif

  #pragma unroll
  for (int j = 0; j < 8; ++j)
    #pragma unroll
    for (int i = 0; i < 8; ++i) {
      long r = (long)tile * 16 + i + 8 * half;   // C/D: VGPR i -> row i / i+8
      OUT[r * DH + j * 16 + l16] = c[j][i];
    }
}

// =====================================================================
// Self-loop attr = mean of edge_attr over incoming edges (PyG 'mean')
// =====================================================================
__global__ void fill_f32(float* __restrict__ p, float v, long n) {
  long t = (long)blockIdx.x * blockDim.x + threadIdx.x;
  if (t < n) p[t] = v;
}

__global__ void loop_accum(const int* __restrict__ ei, const float* __restrict__ ea,
                           float* __restrict__ sums, float* __restrict__ cnts, int E_) {
  int e = blockIdx.x * blockDim.x + threadIdx.x;
  if (e >= E_) return;
  int d = ei[E_ + e];
  atomicAdd(&cnts[d], 1.0f);
  #pragma unroll
  for (int k = 0; k < 4; ++k) atomicAdd(&sums[(size_t)d * 4 + k], ea[(size_t)e * 4 + k]);
}

__global__ void loop_final(const float* __restrict__ sums, const float* __restrict__ cnts,
                           float* __restrict__ lattr, int N_) {
  int n = blockIdx.x * blockDim.x + threadIdx.x;
  if (n >= N_) return;
  float c = fmaxf(cnts[n], 1.0f);
  #pragma unroll
  for (int k = 0; k < 4; ++k) lattr[(size_t)n * 4 + k] = sums[(size_t)n * 4 + k] / c;
}

// =====================================================================
// Per-layer segment-softmax pipeline
// =====================================================================
__global__ void init_seg(float* __restrict__ acc, float* __restrict__ denom,
                         unsigned* __restrict__ emax, int N_) {
  long t = (long)blockIdx.x * blockDim.x + threadIdx.x;
  if (t < (long)N_ * DH) acc[t] = 0.0f;
  if (t < N_) { denom[t] = 0.0f; emax[t] = 0u; }  // 0u == encoding of -max
}

// pass A: per-edge logit e = leakyrelu(xl[s]+xr[d]+ea@We) . att ; atomic seg-max
__global__ __launch_bounds__(256) void edge_logits(
    const int* __restrict__ ei, const float* __restrict__ ea,
    const float* __restrict__ lattr, const float* __restrict__ xl,
    const float* __restrict__ xr, const float* __restrict__ We,
    const float* __restrict__ att, float* __restrict__ ebuf,
    unsigned* __restrict__ emax, int E_, int N_) {
  const int lane = threadIdx.x & 31;
  const int wId  = (int)((blockIdx.x * blockDim.x + threadIdx.x) >> 5);
  const int nW   = (int)((gridDim.x * blockDim.x) >> 5);
  float wek[4][4], attj[4];
  #pragma unroll
  for (int j = 0; j < 4; ++j) {
    attj[j] = att[32 * j + lane];
    #pragma unroll
    for (int k = 0; k < 4; ++k) wek[k][j] = We[(size_t)k * DH + 32 * j + lane];
  }
  const int Etot = E_ + N_;
  for (int e = wId; e < Etot; e += nW) {
    int s, d; const float* ap;
    if (e < E_) { s = ei[e]; d = ei[E_ + e]; ap = ea + (size_t)e * 4; }
    else        { s = d = e - E_;            ap = lattr + (size_t)(e - E_) * 4; }
    const float a0 = ap[0], a1 = ap[1], a2 = ap[2], a3 = ap[3];
    const float* pl = xl + (size_t)s * DH;
    const float* pr = xr + (size_t)d * DH;
    float acc = 0.0f;
    #pragma unroll
    for (int j = 0; j < 4; ++j) {
      int h = 32 * j + lane;
      float m = pl[h] + pr[h] +
                a0 * wek[0][j] + a1 * wek[1][j] + a2 * wek[2][j] + a3 * wek[3][j];
      m = (m > 0.0f) ? m : NEG_SLOPE * m;
      acc += m * attj[j];
    }
    acc = wave_sum32(acc);
    if (lane == 0) {
      ebuf[e] = acc;
      atomicMax(&emax[d], enc_f32(acc));
    }
  }
}

// pass B: ee = exp(e - emax[d]); denom[d] += ee
__global__ void edge_exp(const int* __restrict__ ei, float* __restrict__ ebuf,
                         const unsigned* __restrict__ emax, float* __restrict__ denom,
                         int E_, int N_) {
  int e = blockIdx.x * blockDim.x + threadIdx.x;
  int Etot = E_ + N_;
  if (e >= Etot) return;
  int d = (e < E_) ? ei[E_ + e] : (e - E_);
  float v = __expf(ebuf[e] - dec_f32(emax[d]));
  ebuf[e] = v;
  atomicAdd(&denom[d], v);
}

// pass C: acc[d] += (ee/denom[d]) * xl[s]
__global__ __launch_bounds__(256) void edge_scatter(
    const int* __restrict__ ei, const float* __restrict__ ebuf,
    const float* __restrict__ denom, const float* __restrict__ xl,
    float* __restrict__ acc, int E_, int N_) {
  const int lane = threadIdx.x & 31;
  const int wId  = (int)((blockIdx.x * blockDim.x + threadIdx.x) >> 5);
  const int nW   = (int)((gridDim.x * blockDim.x) >> 5);
  const int Etot = E_ + N_;
  for (int e = wId; e < Etot; e += nW) {
    int s, d;
    if (e < E_) { s = ei[e]; d = ei[E_ + e]; } else { s = d = e - E_; }
    float alpha = ebuf[e] / denom[d];
    const float* pl = xl + (size_t)s * DH;
    float* pd = acc + (size_t)d * DH;
    #pragma unroll
    for (int j = 0; j < 4; ++j) {
      int h = 32 * j + lane;
      atomicAdd(&pd[h], alpha * pl[h]);
    }
  }
}

__global__ void bias_elu(const float* __restrict__ in, const float* __restrict__ b,
                         float* __restrict__ out, long total) {
  long t = (long)blockIdx.x * blockDim.x + threadIdx.x;
  if (t >= total) return;
  float v = in[t] + b[t & (DH - 1)];
  out[t] = (v > 0.0f) ? v : (__expf(v) - 1.0f);
}

// out[n] = (h2[n] + x[n]) . Wfc + bfc
__global__ __launch_bounds__(256) void final_fc(
    const float* __restrict__ h2, const float* __restrict__ x,
    const float* __restrict__ Wfc, const float* __restrict__ bfc,
    float* __restrict__ out, int N_) {
  const int lane = threadIdx.x & 31;
  const int wId  = (int)((blockIdx.x * blockDim.x + threadIdx.x) >> 5);
  const int nW   = (int)((gridDim.x * blockDim.x) >> 5);
  float wv[4];
  #pragma unroll
  for (int j = 0; j < 4; ++j) wv[j] = Wfc[32 * j + lane];
  for (int n = wId; n < N_; n += nW) {
    const float* ph = h2 + (size_t)n * DH;
    const float* px = x + (size_t)n * DH;
    float acc = 0.0f;
    #pragma unroll
    for (int j = 0; j < 4; ++j) {
      int h = 32 * j + lane;
      acc += (ph[h] + px[h]) * wv[j];
    }
    acc = wave_sum32(acc);
    if (lane == 0) out[n] = acc + bfc[0];
  }
}

// =====================================================================
extern "C" void kernel_launch(void* const* d_in, const int* in_sizes, int n_in,
                              void* d_out, int out_size, void* d_ws, size_t ws_size,
                              hipStream_t stream) {
  const float* x    = (const float*)d_in[0];
  const int*   ei   = (const int*)d_in[1];
  const float* ea   = (const float*)d_in[2];
  const float* W1l  = (const float*)d_in[3];
  const float* W1r  = (const float*)d_in[4];
  const float* W1e  = (const float*)d_in[5];
  const float* att1 = (const float*)d_in[6];
  const float* b1   = (const float*)d_in[7];
  const float* W2l  = (const float*)d_in[8];
  const float* W2r  = (const float*)d_in[9];
  const float* W2e  = (const float*)d_in[10];
  const float* att2 = (const float*)d_in[11];
  const float* b2   = (const float*)d_in[12];
  const float* Wfc  = (const float*)d_in[13];
  const float* bfc  = (const float*)d_in[14];
  float* out = (float*)d_out;
  (void)n_in; (void)out_size; (void)ws_size;

  const int Nn   = in_sizes[0] / DH;
  const int Ee   = in_sizes[2] / 4;
  const int Etot = Ee + Nn;

  // workspace layout (floats)
  float* p = (float*)d_ws;
  float*    xl    = p; p += (size_t)Nn * DH;
  float*    xr    = p; p += (size_t)Nn * DH;
  float*    accb  = p; p += (size_t)Nn * DH;
  float*    hbuf  = p; p += (size_t)Nn * DH;
  float*    ebuf  = p; p += (size_t)Etot;
  float*    denom = p; p += (size_t)Nn;
  unsigned* emax  = (unsigned*)p; p += (size_t)Nn;
  float*    sums  = p; p += (size_t)Nn * 4;
  float*    cnts  = p; p += (size_t)Nn;
  float*    lattr = p; p += (size_t)Nn * 4;

  const int TPB = 256;
  const int nTiles   = (Nn + 15) / 16;
  const int gemmBlk  = (nTiles + 7) / 8;
  const int nodeBlk  = (Nn + TPB - 1) / TPB;
  const int featBlk  = (int)(((size_t)Nn * DH + TPB - 1) / TPB);
  const int edgeBlk  = (Etot + TPB - 1) / TPB;
  const int waveGrid = 2048;

  // self-loop mean edge attr (sums and cnts are contiguous -> one fill)
  fill_f32<<<(Nn * 5 + TPB - 1) / TPB, TPB, 0, stream>>>(sums, 0.0f, (long)Nn * 5);
  loop_accum<<<(Ee + TPB - 1) / TPB, TPB, 0, stream>>>(ei, ea, sums, cnts, Ee);
  loop_final<<<nodeBlk, TPB, 0, stream>>>(sums, cnts, lattr, Nn);

  // ---- layer 1 ----
  gemm_x128_wmma<<<gemmBlk, TPB, 0, stream>>>(x, W1l, xl, nTiles);
  gemm_x128_wmma<<<gemmBlk, TPB, 0, stream>>>(x, W1r, xr, nTiles);
  init_seg<<<featBlk, TPB, 0, stream>>>(accb, denom, emax, Nn);
  edge_logits<<<waveGrid, TPB, 0, stream>>>(ei, ea, lattr, xl, xr, W1e, att1, ebuf, emax, Ee, Nn);
  edge_exp<<<edgeBlk, TPB, 0, stream>>>(ei, ebuf, emax, denom, Ee, Nn);
  edge_scatter<<<waveGrid, TPB, 0, stream>>>(ei, ebuf, denom, xl, accb, Ee, Nn);
  bias_elu<<<featBlk, TPB, 0, stream>>>(accb, b1, hbuf, (long)Nn * DH);

  // ---- layer 2 ----
  gemm_x128_wmma<<<gemmBlk, TPB, 0, stream>>>(hbuf, W2l, xl, nTiles);
  gemm_x128_wmma<<<gemmBlk, TPB, 0, stream>>>(hbuf, W2r, xr, nTiles);
  init_seg<<<featBlk, TPB, 0, stream>>>(accb, denom, emax, Nn);
  edge_logits<<<waveGrid, TPB, 0, stream>>>(ei, ea, lattr, xl, xr, W2e, att2, ebuf, emax, Ee, Nn);
  edge_exp<<<edgeBlk, TPB, 0, stream>>>(ei, ebuf, emax, denom, Ee, Nn);
  edge_scatter<<<waveGrid, TPB, 0, stream>>>(ei, ebuf, denom, xl, accb, Ee, Nn);
  bias_elu<<<featBlk, TPB, 0, stream>>>(accb, b2, accb, (long)Nn * DH);

  // ---- residual + fc ----
  final_fc<<<waveGrid, TPB, 0, stream>>>(accb, x, Wfc, bfc, out, Nn);
}
